// HET_RelationalAttLayer_23493471109978
// MI455X (gfx1250) — compile-verified, matrix operand order checked by reading, not run on hardware
//
#include <hip/hip_runtime.h>
#include <hip/hip_bf16.h>

#define N_NODES   100000
#define N_EDGES   1600000
#define N_REL     8
#define N_HEADS   4
#define IN_FEAT   128
#define OUT_FEAT  64
#define D_HEAD    16
#define RH        (N_REL * N_HEADS)      // 32 (relation*head pairs)
#define PROJ_COLS (N_REL * OUT_FEAT)     // 512
#define LEAKY     0.2f

typedef __attribute__((ext_vector_type(8)))  __bf16 v8bf;
typedef __attribute__((ext_vector_type(16))) __bf16 v16bf;
typedef __attribute__((ext_vector_type(8)))  float  v8f;
typedef __attribute__((ext_vector_type(4)))  float  v4f;

// Order-preserving float <-> uint bijection so segment-max can use native
// global_atomic_max_u32 (no CAS loop).
__device__ __forceinline__ unsigned f2ord(float f) {
  unsigned u = __float_as_uint(f);
  return (u & 0x80000000u) ? ~u : (u | 0x80000000u);
}
__device__ __forceinline__ float ord2f(unsigned m) {
  unsigned u = (m & 0x80000000u) ? (m ^ 0x80000000u) : ~m;
  return __uint_as_float(u);
}

// ---------------------------------------------------------------- fp32 -> bf16
__global__ void k_cvt_bf16(const float* __restrict__ x, __bf16* __restrict__ y, int n) {
  int i = blockIdx.x * blockDim.x + threadIdx.x;
  if (i < n) y[i] = (__bf16)x[i];
}

// ------------------------------------------------- pack W into B-fragment order
// conv_weights[R][H][IN][Dh] -> Bpk[(ct*4+kb)*32 + lane][16 bf16], matching the
// 16-bit B-matrix VGPR striping (lane&15 = N, lane>>4 selects K-halves).
__global__ void k_pack_w(const float* __restrict__ w, __bf16* __restrict__ bp) {
  int i = blockIdx.x * 256 + threadIdx.x;        // 32*4*32*16 = 65536 elements
  int e    = i & 15;
  int lane = (i >> 4) & 31;
  int kb   = (i >> 9) & 3;
  int ct   = i >> 11;                            // 0..31 == r*4+h
  int d    = lane & 15;                          // N (output column in tile)
  int k    = kb * 32 + ((lane >> 4) * 8) + ((e < 8) ? 0 : 16) + (e & 7);
  bp[i] = (__bf16)w[(ct * IN_FEAT + k) * D_HEAD + d];
}

// ------------------------------------------------------- projection GEMM (WMMA)
// proj[N,512] = A_bf16[N,128] x W_bf16[128,512]
// 4 waves/block; wave w computes column tiles [8w, 8w+8) of one 16-node block.
__global__ __launch_bounds__(128) void k_proj_wmma(const __bf16* __restrict__ A,
                                                   const __bf16* __restrict__ Bp,
                                                   float* __restrict__ proj) {
  const int lane = threadIdx.x & 31;
  const int wave = threadIdx.x >> 5;
  const int nodeBase = blockIdx.x * 16;
  const int ctBase = wave * 8;
  const int m    = lane & 15;                    // row within A / col within C
  const int half = lane >> 4;

  v8f acc[8];
#pragma unroll
  for (int t = 0; t < 8; ++t) acc[t] = (v8f)0.0f;

  const __bf16* arow = A + (size_t)(nodeBase + m) * IN_FEAT + half * 8;

#pragma unroll
  for (int kb = 0; kb < 4; ++kb) {
    union { v16bf v; v8bf h[2]; } af;            // A frag: K 0-7 & 16-23 (or +8)
    af.h[0] = *(const v8bf*)(arow + kb * 32);
    af.h[1] = *(const v8bf*)(arow + kb * 32 + 16);
#pragma unroll
    for (int t = 0; t < 8; ++t) {
      const int ct = ctBase + t;
      const v16bf bf = *(const v16bf*)(Bp + ((ct * 4 + kb) * 32 + lane) * 16);
      acc[t] = __builtin_amdgcn_wmma_f32_16x16x32_bf16(
          false, af.v, false, bf, (short)0, acc[t], false, false);
    }
  }

  // C/D layout: VGPR j, lanes0-15 -> M=j, lanes16-31 -> M=j+8; N = lane&15.
#pragma unroll
  for (int t = 0; t < 8; ++t) {
    const int ct = ctBase + t;
    float* cp = proj + (size_t)(nodeBase + half * 8) * PROJ_COLS + ct * 16 + m;
#pragma unroll
    for (int j = 0; j < 8; ++j) cp[(size_t)j * PROJ_COLS] = acc[t][j];
  }
}

// -------------------------------------------------- init out=bias, smax, denom
__global__ void k_init(float* __restrict__ out, const float* __restrict__ bias,
                       unsigned* __restrict__ smax, float* __restrict__ denom) {
  int i = blockIdx.x * blockDim.x + threadIdx.x;
  if (i < N_NODES * OUT_FEAT) out[i] = bias[i & (OUT_FEAT - 1)];
  if (i < N_NODES * N_HEADS) { smax[i] = 0u; denom[i] = 0.0f; }
}

// --------------------------------- per-node el/er = <proj[n,rh,:], attn_{l,r}>
__global__ void k_elr(const float* __restrict__ proj,
                      const float* __restrict__ al, const float* __restrict__ ar,
                      float* __restrict__ eln, float* __restrict__ ern) {
  int i = blockIdx.x * blockDim.x + threadIdx.x;   // n*32 + rh
  if (i >= N_NODES * RH) return;
  int rh = i & (RH - 1);
  const float* p = proj + (size_t)(i >> 5) * PROJ_COLS + rh * 16;
  const float* l = al + rh * 16;
  const float* r = ar + rh * 16;
  float sl = 0.0f, sr = 0.0f;
#pragma unroll
  for (int d = 0; d < 16; ++d) { float v = p[d]; sl += v * l[d]; sr += v * r[d]; }
  eln[i] = sl;
  ern[i] = sr;
}

// ---------------------------- per-edge score = leaky(el+er); segment-max atomic
__global__ void k_score(const int* __restrict__ src, const int* __restrict__ dst,
                        const int* __restrict__ rel,
                        const float* __restrict__ eln, const float* __restrict__ ern,
                        float* __restrict__ score, unsigned* __restrict__ smax) {
  int e = blockIdx.x * blockDim.x + threadIdx.x;
  if (e >= N_EDGES) return;
  int s = src[e], d = dst[e], r = rel[e];
  const v4f el = *(const v4f*)(eln + (size_t)s * RH + r * N_HEADS);
  const v4f er = *(const v4f*)(ern + (size_t)d * RH + r * N_HEADS);
  v4f sc;
#pragma unroll
  for (int h = 0; h < N_HEADS; ++h) {
    float v = el[h] + er[h];
    float lv = v > 0.0f ? v : LEAKY * v;
    sc[h] = lv;
    atomicMax(&smax[d * N_HEADS + h], f2ord(lv));   // global_atomic_max_u32
  }
  *(v4f*)(score + (size_t)e * N_HEADS) = sc;
}

// --------------------------------------- e = exp(score - smax[dst]); sum denom
__global__ void k_exp(const int* __restrict__ dst, float* __restrict__ score,
                      const unsigned* __restrict__ smax, float* __restrict__ denom) {
  int e = blockIdx.x * blockDim.x + threadIdx.x;
  if (e >= N_EDGES) return;
  int d = dst[e];
  v4f sc = *(const v4f*)(score + (size_t)e * N_HEADS);
#pragma unroll
  for (int h = 0; h < N_HEADS; ++h) {
    float ex = __expf(sc[h] - ord2f(smax[d * N_HEADS + h]));
    sc[h] = ex;
    unsafeAtomicAdd(&denom[d * N_HEADS + h], ex);   // global_atomic_add_f32
  }
  *(v4f*)(score + (size_t)e * N_HEADS) = sc;
}

// ---------------------------------------------------- alpha = e / denom[dst]
__global__ void k_alpha(const int* __restrict__ dst, float* __restrict__ score,
                        const float* __restrict__ denom) {
  int i = blockIdx.x * blockDim.x + threadIdx.x;    // e*4 + h
  if (i >= N_EDGES * N_HEADS) return;
  int e = i >> 2, h = i & 3;
  score[i] = score[i] / denom[dst[e] * N_HEADS + h];
}

// ---------------------- out[dst] += alpha * proj[src, rel]  (64 feats / edge)
__global__ void k_agg(const int* __restrict__ src, const int* __restrict__ dst,
                      const int* __restrict__ rel, const float* __restrict__ alpha,
                      const float* __restrict__ proj, float* __restrict__ out) {
  int i = blockIdx.x * blockDim.x + threadIdx.x;    // edge*64 + f
  int f = i & (OUT_FEAT - 1);
  int e = i >> 6;
  int s = src[e], d = dst[e], r = rel[e];
  float a = alpha[(size_t)e * N_HEADS + (f >> 4)];
  unsafeAtomicAdd(&out[(size_t)d * OUT_FEAT + f],
                  a * proj[(size_t)s * PROJ_COLS + r * OUT_FEAT + f]);
}

extern "C" void kernel_launch(void* const* d_in, const int* in_sizes, int n_in,
                              void* d_out, int out_size, void* d_ws, size_t ws_size,
                              hipStream_t stream) {
  const float* inputs  = (const float*)d_in[0];   // [N,128]
  const float* weights = (const float*)d_in[1];   // [8,4,128,16]
  const float* attn_l  = (const float*)d_in[2];   // [8,4,16]
  const float* attn_r  = (const float*)d_in[3];   // [8,4,16]
  const float* h_bias  = (const float*)d_in[4];   // [64]
  const int*   src_idx = (const int*)d_in[5];     // [E]
  const int*   dst_idx = (const int*)d_in[6];     // [E]
  const int*   rel_idx = (const int*)d_in[7];     // [E]
  float* out = (float*)d_out;                     // [N,64]

  // Workspace carve-up (all chunks are 256B-aligned sizes).
  char* ws = (char*)d_ws;
  __bf16*   Abf   = (__bf16*)ws;                               ws += (size_t)N_NODES * IN_FEAT * 2;     // 25.6 MB
  __bf16*   Bpk   = (__bf16*)ws;                               ws += (size_t)IN_FEAT * PROJ_COLS * 2;   // 128 KB
  float*    proj  = (float*)ws;                                ws += (size_t)N_NODES * PROJ_COLS * 4;   // 204.8 MB
  float*    eln   = (float*)ws;                                ws += (size_t)N_NODES * RH * 4;          // 12.8 MB
  float*    ern   = (float*)ws;                                ws += (size_t)N_NODES * RH * 4;          // 12.8 MB
  float*    score = (float*)ws;                                ws += (size_t)N_EDGES * N_HEADS * 4;     // 25.6 MB
  unsigned* smax  = (unsigned*)ws;                             ws += (size_t)N_NODES * N_HEADS * 4;     // 1.6 MB
  float*    denom = (float*)ws;                                ws += (size_t)N_NODES * N_HEADS * 4;     // 1.6 MB

  const int B = 256;

  // 1) fp32 -> bf16 activations
  k_cvt_bf16<<<(N_NODES * IN_FEAT) / B, B, 0, stream>>>(inputs, Abf, N_NODES * IN_FEAT);
  // 2) pack weights into WMMA B-fragment order
  k_pack_w<<<(IN_FEAT * PROJ_COLS) / B, B, 0, stream>>>(weights, Bpk);
  // 3) projection GEMM via v_wmma_f32_16x16x32_bf16
  k_proj_wmma<<<N_NODES / 16, 128, 0, stream>>>(Abf, Bpk, proj);
  // 4) init out = bias, smax = -inf(ordered 0), denom = 0
  k_init<<<(N_NODES * OUT_FEAT) / B, B, 0, stream>>>(out, h_bias, smax, denom);
  // 5) per-node el/er reductions
  k_elr<<<(N_NODES * RH) / B, B, 0, stream>>>(proj, attn_l, attn_r, eln, ern);
  // 6) per-edge scores + segment max (native u32 atomic max)
  k_score<<<(N_EDGES + B - 1) / B, B, 0, stream>>>(src_idx, dst_idx, rel_idx, eln, ern, score, smax);
  // 7) exp + segment sum (native f32 atomic add)
  k_exp<<<(N_EDGES + B - 1) / B, B, 0, stream>>>(dst_idx, score, smax, denom);
  // 8) normalize to alpha
  k_alpha<<<(N_EDGES * N_HEADS) / B, B, 0, stream>>>(dst_idx, score, denom);
  // 9) weighted aggregation into out
  k_agg<<<(N_EDGES * OUT_FEAT) / B, B, 0, stream>>>(src_idx, dst_idx, rel_idx, score, proj, out);
}
